// VectorQuantizer_73117523247499
// MI455X (gfx1250) — matrix operand order repaired and checked by reading, not compiled, not run on hardware
//
#include <hip/hip_runtime.h>

#define NTOK 8192
#define NCODE 8192
#define DIM 512
#define SK_EPS 0.003f
#define LOGN 9.0109133472792881f   // log(8192)

typedef __attribute__((ext_vector_type(16))) __bf16 v16bf;
typedef __attribute__((ext_vector_type(8)))  float  v8f;
typedef __attribute__((ext_vector_type(8)))  unsigned short us8;
typedef __attribute__((ext_vector_type(16))) unsigned short us16;
typedef __attribute__((ext_vector_type(4)))  unsigned int   u32x4;
typedef __attribute__((ext_vector_type(8)))  int            i32x8;
typedef __attribute__((ext_vector_type(4)))  int            i32x4;

// ---------------------------------------------------------------------------
// K1: row sums-of-squares for x and emb, plus fp32 -> bf16 (RNE) conversion
// ---------------------------------------------------------------------------
__global__ __launch_bounds__(256)
void k_prep(const float* __restrict__ x, const float* __restrict__ emb,
            unsigned short* __restrict__ Xh, unsigned short* __restrict__ Eh,
            float* __restrict__ xs, float* __restrict__ es)
{
    __shared__ float red[256];
    int row = blockIdx.x;
    const float* src; unsigned short* dst; float* sq; int r;
    if (row < NTOK) { src = x;   dst = Xh; sq = xs; r = row; }
    else            { src = emb; dst = Eh; sq = es; r = row - NTOK; }
    const float* p = src + (size_t)r * DIM;
    float acc = 0.f;
    for (int j = threadIdx.x; j < DIM; j += 256) {
        float v = p[j];
        acc += v * v;
        unsigned u  = __float_as_uint(v);
        unsigned rr = u + 0x7FFFu + ((u >> 16) & 1u);   // round-to-nearest-even
        dst[(size_t)r * DIM + j] = (unsigned short)(rr >> 16);
    }
    red[threadIdx.x] = acc;
    __syncthreads();
    for (int s = 128; s > 0; s >>= 1) {
        if (threadIdx.x < s) red[threadIdx.x] += red[threadIdx.x + s];
        __syncthreads();
    }
    if (threadIdx.x == 0) sq[r] = red[0];
}

// ---------------------------------------------------------------------------
// K2: S = X * E^T via bf16 WMMA, f32 accumulate, stored fp16 (128 MB -> L2).
// LDS tiles are filled by the Tensor Data Mover (TDM): one tensor_load_to_lds
// per 32x128 bf16 tile, issued by wave 0, tracked with TENSORcnt.
// D# pad feature reproduces the +8-element (16 B) row pad:
//   pad_interval=3 (every 16 DWORDs = one 64 B tile row), pad_amount=3 (4 DW).
// Block = 256 threads (8 waves), 128x128 output tile, BK = 32.
// Wave grid 2x4: each wave owns a 64(M) x 32(N) region = 4x2 WMMA tiles.
// A-frag (16x32 bf16): lane<16 -> M=lane, K {0..7,16..23}; lane>=16 -> K {8..15,24..31}
// B-frag (32x16 bf16): lane<16 -> N=lane, K {0..7,16..23}; lane>=16 -> K {8..15,24..31}
// ---------------------------------------------------------------------------
__global__ __launch_bounds__(256)
void k_gemm(const unsigned short* __restrict__ Xh, const unsigned short* __restrict__ Eh,
            _Float16* __restrict__ S)
{
    __shared__ unsigned short Xs[128 * 40];   // 40 = 32 + 8 bf16 pad per row
    __shared__ unsigned short Es[128 * 40];
    const int tid  = threadIdx.x;
    const int lane = tid & 31;
    const int wid  = tid >> 5;
    const int mW   = (wid & 1) * 64;
    const int nW   = (wid >> 1) * 32;
    const int bBase = blockIdx.y * 128;   // token rows  (M)
    const int nBase = blockIdx.x * 128;   // code  rows  (N)
    const int hsel = lane >> 4, lm = lane & 15;

    v8f acc[4][2] = {};

    // ---- TDM descriptor group 1 (tile/tensor geometry, constant over k) ----
    // [17:16] data_size=1 (2B) | [20] pad_enable | [24:22] pad_interval=3
    // | [31:25] pad_amount=3 ; tensor_dim0=512 ; tensor_dim1=8192 ;
    // tile_dim0=32 ; tile_dim1=128 ; tensor_dim0_stride=512
    const i32x8 g1 = { (int)0x06D10000u,
                       (int)(512u  << 16),   // tensor_dim0 lo16 @ [31:16]
                       (int)(8192u << 16),   // dim0 hi16=0 | tensor_dim1 lo16
                       (int)(32u   << 16),   // dim1 hi16=0 | tile_dim0
                       128,                  // tile_dim1 | tile_dim2=0
                       512,                  // tensor_dim0_stride lo32
                       0, 0 };
    const i32x4 gz4 = { 0, 0, 0, 0 };
    const i32x8 gz8 = { 0, 0, 0, 0, 0, 0, 0, 0 };
    const unsigned ldsX = (unsigned)(size_t)(void*)Xs;
    const unsigned ldsE = (unsigned)(size_t)(void*)Es;
    const unsigned long long gaX0 =
        (unsigned long long)(size_t)(Xh + (size_t)bBase * DIM);
    const unsigned long long gaE0 =
        (unsigned long long)(size_t)(Eh + (size_t)nBase * DIM);

    for (int kt = 0; kt < DIM; kt += 32) {
        __syncthreads();              // previous iteration's consumers done
        if (wid == 0) {
            unsigned long long gaX = gaX0 + (unsigned)kt * 2u;
            unsigned long long gaE = gaE0 + (unsigned)kt * 2u;
            u32x4 g0x, g0e;
            g0x[0] = 1u;  g0x[1] = ldsX;
            g0x[2] = (unsigned)gaX;
            g0x[3] = ((unsigned)(gaX >> 32) & 0x01FFFFFFu) | 0x80000000u; // type=2
            g0e[0] = 1u;  g0e[1] = ldsE;
            g0e[2] = (unsigned)gaE;
            g0e[3] = ((unsigned)(gaE >> 32) & 0x01FFFFFFu) | 0x80000000u;
            __builtin_amdgcn_tensor_load_to_lds(g0x, g1, gz4, gz4, gz8, 0);
            __builtin_amdgcn_tensor_load_to_lds(g0e, g1, gz4, gz4, gz8, 0);
            __builtin_amdgcn_s_wait_tensorcnt(0);
        }
        __syncthreads();              // tile visible to all waves

        v16bf afr[4], bfr[2];
        const int kbA = hsel * 8;
        #pragma unroll
        for (int mt = 0; mt < 4; ++mt) {
            const unsigned short* pr = &Xs[(mW + mt * 16 + lm) * 40];
            us8 lo = *(const us8*)(pr + kbA);
            us8 hi = *(const us8*)(pr + kbA + 16);
            us16 t = __builtin_shufflevector(lo, hi, 0,1,2,3,4,5,6,7,8,9,10,11,12,13,14,15);
            afr[mt] = __builtin_bit_cast(v16bf, t);
        }
        const int koB = hsel * 8;
        #pragma unroll
        for (int nt = 0; nt < 2; ++nt) {
            const unsigned short* pr = &Es[(nW + nt * 16 + lm) * 40 + koB];
            us8 lo = *(const us8*)pr;
            us8 hi = *(const us8*)(pr + 16);
            us16 t = __builtin_shufflevector(lo, hi, 0,1,2,3,4,5,6,7,8,9,10,11,12,13,14,15);
            bfr[nt] = __builtin_bit_cast(v16bf, t);
        }
        #pragma unroll
        for (int mt = 0; mt < 4; ++mt)
            #pragma unroll
            for (int nt = 0; nt < 2; ++nt)
                acc[mt][nt] = __builtin_amdgcn_wmma_f32_16x16x32_bf16(
                    false, afr[mt], false, bfr[nt], (short)0, acc[mt][nt], false, false);
    }

    // C/D layout: lane (hsel,lm), VGPR j -> (M = hsel*8 + j, N = lm)
    #pragma unroll
    for (int mt = 0; mt < 4; ++mt)
        #pragma unroll
        for (int nt = 0; nt < 2; ++nt)
            #pragma unroll
            for (int j = 0; j < 8; ++j) {
                int rowg = bBase + mW + mt * 16 + hsel * 8 + j;
                int colg = nBase + nW + nt * 16 + lm;
                S[(size_t)rowg * NCODE + colg] = (_Float16)acc[mt][nt][j];
            }
}

// ---------------------------------------------------------------------------
// K3: streaming min/max of d = xs[b] + es[k] - 2*S  -> per-block partials
// ---------------------------------------------------------------------------
__global__ __launch_bounds__(256)
void k_minmax(const _Float16* __restrict__ S, const float* __restrict__ xs,
              const float* __restrict__ es, float* __restrict__ part)
{
    __shared__ float rmin[256], rmax[256];
    float mn = 3.4e38f, mx = -3.4e38f;
    const size_t total = (size_t)NTOK * NCODE;
    for (size_t i = (size_t)blockIdx.x * 256 + threadIdx.x; i < total;
         i += (size_t)gridDim.x * 256) {
        int b = (int)(i >> 13);
        int k = (int)(i & 8191);
        float d = xs[b] + es[k] - 2.f * (float)S[i];
        mn = fminf(mn, d); mx = fmaxf(mx, d);
    }
    rmin[threadIdx.x] = mn; rmax[threadIdx.x] = mx;
    __syncthreads();
    for (int s = 128; s > 0; s >>= 1) {
        if (threadIdx.x < s) {
            rmin[threadIdx.x] = fminf(rmin[threadIdx.x], rmin[threadIdx.x + s]);
            rmax[threadIdx.x] = fmaxf(rmax[threadIdx.x], rmax[threadIdx.x + s]);
        }
        __syncthreads();
    }
    if (threadIdx.x == 0) { part[2 * blockIdx.x] = rmin[0]; part[2 * blockIdx.x + 1] = rmax[0]; }
}

// K4: finalize constants (middle, invA = 1/(eps*amplitude)) and zero lc
__global__ __launch_bounds__(256)
void k_finalize(const float* __restrict__ part, float* __restrict__ consts,
                float* __restrict__ lc)
{
    __shared__ float rmin[256], rmax[256];
    float mn = 3.4e38f, mx = -3.4e38f;
    for (int i = threadIdx.x; i < 4096; i += 256) {
        mn = fminf(mn, part[2 * i]); mx = fmaxf(mx, part[2 * i + 1]);
    }
    rmin[threadIdx.x] = mn; rmax[threadIdx.x] = mx;
    __syncthreads();
    for (int s = 128; s > 0; s >>= 1) {
        if (threadIdx.x < s) {
            rmin[threadIdx.x] = fminf(rmin[threadIdx.x], rmin[threadIdx.x + s]);
            rmax[threadIdx.x] = fmaxf(rmax[threadIdx.x], rmax[threadIdx.x + s]);
        }
        __syncthreads();
    }
    if (threadIdx.x == 0) {
        float middle = (rmax[0] + rmin[0]) * 0.5f;
        float amp = rmax[0] - middle + 1e-5f;
        consts[0] = middle;
        consts[1] = 1.f / (SK_EPS * amp);
    }
    for (int i = threadIdx.x; i < NTOK; i += 256) lc[i] = 0.f;
}

// ---------------------------------------------------------------------------
// K5: row (code) update partials: online LSE over a 512-token chunk per k
// ---------------------------------------------------------------------------
__global__ __launch_bounds__(256)
void k_row(const _Float16* __restrict__ S, const float* __restrict__ xs,
           const float* __restrict__ lc, const float* __restrict__ consts,
           float* __restrict__ pm, float* __restrict__ ps)
{
    int k = blockIdx.x * 256 + threadIdx.x;
    int c = blockIdx.y;
    float middle = consts[0], invA = consts[1];
    float m = -3.4e38f, s = 0.f;
    int b0 = c * (NTOK / 16);
    for (int b = b0; b < b0 + NTOK / 16; ++b) {
        float u = (middle - xs[b]) * invA + lc[b];
        float v = u + 2.f * invA * (float)S[(size_t)b * NCODE + k];
        if (v > m) { s = s * __expf(m - v) + 1.f; m = v; }
        else       { s += __expf(v - m); }
    }
    pm[(size_t)c * NCODE + k] = m;
    ps[(size_t)c * NCODE + k] = s;
}

// K6: combine 16 chunk-LSE pairs -> lr[k]
__global__ __launch_bounds__(256)
void k_rowfin(const float* __restrict__ pm, const float* __restrict__ ps,
              const float* __restrict__ es, const float* __restrict__ consts,
              float* __restrict__ lr)
{
    int k = blockIdx.x * 256 + threadIdx.x;
    float invA = consts[1];
    float m = -3.4e38f, s = 0.f;
    for (int c = 0; c < 16; ++c) {
        float mc = pm[(size_t)c * NCODE + k];
        float sc = ps[(size_t)c * NCODE + k];
        float M = fmaxf(m, mc);
        s = s * __expf(m - M) + sc * __expf(mc - M);
        m = M;
    }
    lr[k] = es[k] * invA - LOGN - (m + __logf(s));
}

// ---------------------------------------------------------------------------
// K7: column (token) update: one wave per token row, wave-reduced online LSE
// ---------------------------------------------------------------------------
__global__ __launch_bounds__(256)
void k_col(const _Float16* __restrict__ S, const float* __restrict__ xs,
           const float* __restrict__ es, const float* __restrict__ lr,
           const float* __restrict__ consts, float* __restrict__ lc)
{
    int wid = threadIdx.x >> 5, lane = threadIdx.x & 31;
    int b = blockIdx.x * 8 + wid;
    float middle = consts[0], invA = consts[1];
    const _Float16* row = S + (size_t)b * NCODE;
    float m = -3.4e38f, s = 0.f;
    for (int k = lane; k < NCODE; k += 32) {
        float v = lr[k] - es[k] * invA + 2.f * invA * (float)row[k];
        if (v > m) { s = s * __expf(m - v) + 1.f; m = v; }
        else       { s += __expf(v - m); }
    }
    #pragma unroll
    for (int off = 16; off > 0; off >>= 1) {
        float m2 = __shfl_xor(m, off, 32);
        float s2 = __shfl_xor(s, off, 32);
        float M = fmaxf(m, m2);
        s = s * __expf(m - M) + s2 * __expf(m2 - M);
        m = M;
    }
    if (lane == 0)
        lc[b] = -LOGN - ((middle - xs[b]) * invA + m + __logf(s));
}

// ---------------------------------------------------------------------------
// K8: per-token argmax_k (lr[k] - es[k]*invA + 2*invA*S), first-max tiebreak
// ---------------------------------------------------------------------------
__global__ __launch_bounds__(256)
void k_argmax(const _Float16* __restrict__ S, const float* __restrict__ es,
              const float* __restrict__ lr, const float* __restrict__ consts,
              int* __restrict__ idxArr, float* __restrict__ outIdx)
{
    int wid = threadIdx.x >> 5, lane = threadIdx.x & 31;
    int b = blockIdx.x * 8 + wid;
    float invA = consts[1];
    const _Float16* row = S + (size_t)b * NCODE;
    float best = -3.4e38f; int bi = 0;
    for (int k = lane; k < NCODE; k += 32) {
        float v = lr[k] - es[k] * invA + 2.f * invA * (float)row[k];
        if (v > best) { best = v; bi = k; }
    }
    #pragma unroll
    for (int off = 16; off > 0; off >>= 1) {
        float v2 = __shfl_xor(best, off, 32);
        int   i2 = __shfl_xor(bi, off, 32);
        if (v2 > best || (v2 == best && i2 < bi)) { best = v2; bi = i2; }
    }
    if (lane == 0) { idxArr[b] = bi; outIdx[b] = (float)bi; }
}

// K9: gather x_q = emb[idx], per-token SSE partial
__global__ __launch_bounds__(256)
void k_gather(const float* __restrict__ x, const float* __restrict__ emb,
              const int* __restrict__ idxArr, float* __restrict__ xq,
              float* __restrict__ lossPart)
{
    __shared__ float red[256];
    int b = blockIdx.x;
    int idx = idxArr[b];
    const float* e  = emb + (size_t)idx * DIM;
    const float* xr = x + (size_t)b * DIM;
    float acc = 0.f;
    for (int j = threadIdx.x; j < DIM; j += 256) {
        float ev = e[j];
        float d = ev - xr[j];
        xq[(size_t)b * DIM + j] = ev;
        acc += d * d;
    }
    red[threadIdx.x] = acc;
    __syncthreads();
    for (int s = 128; s > 0; s >>= 1) {
        if (threadIdx.x < s) red[threadIdx.x] += red[threadIdx.x + s];
        __syncthreads();
    }
    if (threadIdx.x == 0) lossPart[b] = red[0];
}

// K10: loss = (1 + beta) * mean((x_q - x)^2)
__global__ __launch_bounds__(256)
void k_loss(const float* __restrict__ lossPart, float* __restrict__ out)
{
    __shared__ float red[256];
    float a = 0.f;
    for (int i = threadIdx.x; i < NTOK; i += 256) a += lossPart[i];
    red[threadIdx.x] = a;
    __syncthreads();
    for (int s = 128; s > 0; s >>= 1) {
        if (threadIdx.x < s) red[threadIdx.x] += red[threadIdx.x + s];
        __syncthreads();
    }
    if (threadIdx.x == 0)
        out[0] = 1.25f * red[0] / (float)((size_t)NTOK * DIM);
}

// ---------------------------------------------------------------------------
extern "C" void kernel_launch(void* const* d_in, const int* in_sizes, int n_in,
                              void* d_out, int out_size, void* d_ws, size_t ws_size,
                              hipStream_t stream)
{
    (void)in_sizes; (void)n_in; (void)out_size; (void)ws_size;
    const float* x   = (const float*)d_in[0];
    const float* emb = (const float*)d_in[1];
    float* out = (float*)d_out;

    char* w = (char*)d_ws;
    size_t o = 0;
    auto alloc = [&](size_t bytes) {
        size_t r = o; o += (bytes + 255) & ~(size_t)255; return r;
    };
    _Float16*       S    = (_Float16*)      (w + alloc((size_t)NTOK * NCODE * 2)); // 128 MB
    unsigned short* Xh   = (unsigned short*)(w + alloc((size_t)NTOK * DIM * 2));
    unsigned short* Eh   = (unsigned short*)(w + alloc((size_t)NCODE * DIM * 2));
    float*          xs   = (float*)(w + alloc(NTOK * 4));
    float*          es   = (float*)(w + alloc(NCODE * 4));
    float*          part = (float*)(w + alloc(4096 * 2 * 4));
    float*          cst  = (float*)(w + alloc(64));
    float*          lc   = (float*)(w + alloc(NTOK * 4));
    float*          lr   = (float*)(w + alloc(NCODE * 4));
    float*          pm   = (float*)(w + alloc((size_t)16 * NCODE * 4));
    float*          ps   = (float*)(w + alloc((size_t)16 * NCODE * 4));
    int*            idxA = (int*)  (w + alloc(NTOK * 4));
    float*          lpar = (float*)(w + alloc(NTOK * 4));

    k_prep<<<NTOK + NCODE, 256, 0, stream>>>(x, emb, Xh, Eh, xs, es);
    k_gemm<<<dim3(NCODE / 128, NTOK / 128), 256, 0, stream>>>(Xh, Eh, S);
    k_minmax<<<4096, 256, 0, stream>>>(S, xs, es, part);
    k_finalize<<<1, 256, 0, stream>>>(part, cst, lc);

    for (int it = 0; it < 5; ++it) {
        k_row<<<dim3(NCODE / 256, 16), 256, 0, stream>>>(S, xs, lc, cst, pm, ps);
        k_rowfin<<<NCODE / 256, 256, 0, stream>>>(pm, ps, es, cst, lr);
        if (it < 4)
            k_col<<<NTOK / 8, 256, 0, stream>>>(S, xs, es, lr, cst, lc);
    }

    float* outIdx  = out + (size_t)NTOK * DIM + 1;
    float* outLoss = out + (size_t)NTOK * DIM;
    k_argmax<<<NTOK / 8, 256, 0, stream>>>(S, es, lr, cst, idxA, outIdx);
    k_gather<<<NTOK, 256, 0, stream>>>(x, emb, idxA, out, lpar);
    k_loss<<<1, 256, 0, stream>>>(lpar, outLoss);
}
